// ScannedRNN_27822798143946
// MI455X (gfx1250) — matrix operand order repaired and checked by reading, not compiled, not run on hardware
//
#include <hip/hip_runtime.h>
#include <hip/hip_bf16.h>
#include <math.h>

// ---------------------------------------------------------------------------
// GRU scan: T=512, B=256, D=512, H=512.
//   Phase A (1 launch): pack Wi [D,3H] and [Whrz|Whn] [H,3H] into bf16,
//     pre-swizzled into the CDNA5 WMMA B-matrix (32x16 K-major) lane layout.
//   Phase B (512 chained launches): per timestep, each wave computes a
//     16(batch) x 16(hidden) output block: 4 f32 accumulators fed by
//     v_wmma_f32_16x16x32_bf16, then fused sigmoid/tanh gate math.
// ---------------------------------------------------------------------------

typedef __attribute__((ext_vector_type(16))) __bf16 v16bf;
typedef __attribute__((ext_vector_type(8)))  float  v8f;

#define T_STEPS 512
#define B_DIM   256
#define D_DIM   512
#define H_DIM   512
#define NCOLS   (3 * H_DIM)              // 1536 combined gate columns
#define KTILES  (D_DIM / 32)             // 16 K-tiles per GEMM (D == H)
#define TILE_HALFS 512                   // halfs in one 32x16 B tile
#define W_HALFS ((NCOLS / 16) * KTILES * TILE_HALFS)   // 786432 per weight

__device__ __forceinline__ __bf16 to_bf16(float f) { return (__bf16)f; }

// ---------------------------------------------------------------------------
// Phase A: pack fp32 weights -> bf16 in WMMA B-fragment order.
// B-fragment order used here (contiguous k, same as A): lane L holds column n = L%16; its 16
// halfs hold k = (L/16)*16 + e.  Tile (k0,n0) stored contiguously so the step
// kernel fetches a fragment with two coalesced 128-bit loads per lane.
// ---------------------------------------------------------------------------
__global__ void pack_weights_kernel(const float* __restrict__ Wi,
                                    const float* __restrict__ Whrz,
                                    const float* __restrict__ Whn,
                                    __bf16* __restrict__ pwWi,
                                    __bf16* __restrict__ pwWh) {
    const int total = 2 * W_HALFS;
    for (int i = blockIdx.x * blockDim.x + threadIdx.x; i < total;
         i += gridDim.x * blockDim.x) {
        const int w       = i / W_HALFS;          // 0 = Wi, 1 = Wh-combined
        const int rem     = i - w * W_HALFS;
        const int tileIdx = rem >> 9;             // /512 halfs per tile
        const int within  = rem & 511;
        const int lane    = within >> 4;
        const int e       = within & 15;
        const int n0      = (tileIdx / KTILES) * 16;
        const int k0      = (tileIdx % KTILES) * 32;
        const int k       = k0 + (lane >> 4) * 16 + e;
        const int n       = n0 + (lane & 15);
        float v;
        if (w == 0) {
            v = Wi[(size_t)k * NCOLS + n];
        } else if (n < 2 * H_DIM) {
            v = Whrz[(size_t)k * (2 * H_DIM) + n];
        } else {
            v = Whn[(size_t)k * H_DIM + (n - 2 * H_DIM)];
        }
        (w == 0 ? pwWi : pwWh)[rem] = to_bf16(v);
    }
}

// ---------------------------------------------------------------------------
// Phase B: one GRU timestep.  grid = (H/16, B/16), block = 1 wave (32 lanes).
// ---------------------------------------------------------------------------
__global__ void __launch_bounds__(32)
gru_step_kernel(const float* __restrict__ x_t,            // [B, D]
                const unsigned char* __restrict__ rst_t,  // [B] bool
                const float* __restrict__ h_prev,         // [B, H]
                const float* __restrict__ bi,             // [3H]
                const float* __restrict__ bhn,            // [H]
                const __bf16* __restrict__ pwWi,
                const __bf16* __restrict__ pwWh,
                float* __restrict__ h_out,                // ys[t] : [B, H]
                float* __restrict__ final_h)              // non-null on last t
{
    const int lane = threadIdx.x;            // 0..31
    const int cg   = blockIdx.x * 16;        // hidden column group
    const int rb   = blockIdx.y * 16;        // batch row block

    __shared__ float lds[16][33];            // padded: conflict-free A reads

    v8f acc_r  = {};   // i_r + h@Whrz[:, :H]
    v8f acc_z  = {};   // i_z + h@Whrz[:, H:]
    v8f acc_xn = {};   // i_n           (x part of n-gate)
    v8f acc_hn = {};   // h@Whn         (gets *r before tanh)

    // A-fragment addressing (16-bit A 16x32): lane -> M = lane%16,
// element e -> k = (lane/16)*16 + e
    const int am    = lane & 15;
    const int kbase = (lane >> 4) * 16;

    const int stg_row = lane >> 1;           // staging: 2 lanes per row
    const int stg_col = (lane & 1) * 16;     // each lane copies 16 floats

    // ---------------- x @ Wi : accumulate into r / z / xn -----------------
    for (int k0 = 0; k0 < D_DIM; k0 += 32) {
        __syncthreads();
        {
            const float* src =
                x_t + (size_t)(rb + stg_row) * D_DIM + k0 + stg_col;
            #pragma unroll
            for (int j = 0; j < 16; j += 4) {
                float4 v = *(const float4*)(src + j);
                lds[stg_row][stg_col + j + 0] = v.x;
                lds[stg_row][stg_col + j + 1] = v.y;
                lds[stg_row][stg_col + j + 2] = v.z;
                lds[stg_row][stg_col + j + 3] = v.w;
            }
        }
        __syncthreads();

        v16bf a;
        #pragma unroll
        for (int e = 0; e < 16; ++e) {
            const int k = kbase + e;
            a[e] = to_bf16(lds[am][k]);
        }

        const int kt = k0 >> 5;
        const v16bf br = *(const v16bf*)(
            pwWi + ((size_t)((cg) >> 4) * KTILES + kt) * TILE_HALFS + lane * 16);
        const v16bf bz = *(const v16bf*)(
            pwWi + ((size_t)((H_DIM + cg) >> 4) * KTILES + kt) * TILE_HALFS + lane * 16);
        const v16bf bn = *(const v16bf*)(
            pwWi + ((size_t)((2 * H_DIM + cg) >> 4) * KTILES + kt) * TILE_HALFS + lane * 16);

        acc_r  = __builtin_amdgcn_wmma_f32_16x16x32_bf16(false, a, false, br,
                                                         (short)0, acc_r,  false, false);
        acc_z  = __builtin_amdgcn_wmma_f32_16x16x32_bf16(false, a, false, bz,
                                                         (short)0, acc_z,  false, false);
        acc_xn = __builtin_amdgcn_wmma_f32_16x16x32_bf16(false, a, false, bn,
                                                         (short)0, acc_xn, false, false);
    }

    // ------------- h @ [Whrz|Whn] : accumulate into r / z / hn ------------
    for (int k0 = 0; k0 < H_DIM; k0 += 32) {
        __syncthreads();
        {
            const bool rz = rst_t[rb + stg_row] != 0;   // reset-mask the carry
            const float* src =
                h_prev + (size_t)(rb + stg_row) * H_DIM + k0 + stg_col;
            #pragma unroll
            for (int j = 0; j < 16; j += 4) {
                float4 v;
                if (rz) { v.x = v.y = v.z = v.w = 0.0f; }
                else    { v = *(const float4*)(src + j); }
                lds[stg_row][stg_col + j + 0] = v.x;
                lds[stg_row][stg_col + j + 1] = v.y;
                lds[stg_row][stg_col + j + 2] = v.z;
                lds[stg_row][stg_col + j + 3] = v.w;
            }
        }
        __syncthreads();

        v16bf a;
        #pragma unroll
        for (int e = 0; e < 16; ++e) {
            const int k = kbase + e;
            a[e] = to_bf16(lds[am][k]);
        }

        const int kt = k0 >> 5;
        const v16bf br = *(const v16bf*)(
            pwWh + ((size_t)((cg) >> 4) * KTILES + kt) * TILE_HALFS + lane * 16);
        const v16bf bz = *(const v16bf*)(
            pwWh + ((size_t)((H_DIM + cg) >> 4) * KTILES + kt) * TILE_HALFS + lane * 16);
        const v16bf bn = *(const v16bf*)(
            pwWh + ((size_t)((2 * H_DIM + cg) >> 4) * KTILES + kt) * TILE_HALFS + lane * 16);

        acc_r  = __builtin_amdgcn_wmma_f32_16x16x32_bf16(false, a, false, br,
                                                         (short)0, acc_r,  false, false);
        acc_z  = __builtin_amdgcn_wmma_f32_16x16x32_bf16(false, a, false, bz,
                                                         (short)0, acc_z,  false, false);
        acc_hn = __builtin_amdgcn_wmma_f32_16x16x32_bf16(false, a, false, bn,
                                                         (short)0, acc_hn, false, false);
    }

    // ------------------------- fused gate math ----------------------------
    // C/D layout: element v, lane L -> M = (L/16)*8 + v, N = L%16.
    const int n     = lane & 15;
    const int mrow0 = (lane >> 4) * 8;
    const float b_r  = bi[cg + n];
    const float b_z  = bi[H_DIM + cg + n];
    const float b_n  = bi[2 * H_DIM + cg + n];
    const float b_hn = bhn[cg + n];

    #pragma unroll
    for (int v = 0; v < 8; ++v) {
        const int row = rb + mrow0 + v;
        const float hv = (rst_t[row] != 0)
                             ? 0.0f
                             : h_prev[(size_t)row * H_DIM + cg + n];
        const float r  = 1.0f / (1.0f + __expf(-(acc_r[v] + b_r)));
        const float z  = 1.0f / (1.0f + __expf(-(acc_z[v] + b_z)));
        const float nn = tanhf(acc_xn[v] + b_n + r * (acc_hn[v] + b_hn));
        const float nh = (1.0f - z) * nn + z * hv;
        h_out[(size_t)row * H_DIM + cg + n] = nh;
        if (final_h) final_h[(size_t)row * H_DIM + cg + n] = nh;
    }
}

// ---------------------------------------------------------------------------
// Host driver: 1 pack launch + 512 chained step launches (graph-capturable).
// d_out = [final_h (B*H) | ys (T*B*H)]; ys doubles as the h state chain.
// ---------------------------------------------------------------------------
extern "C" void kernel_launch(void* const* d_in, const int* in_sizes, int n_in,
                              void* d_out, int out_size, void* d_ws, size_t ws_size,
                              hipStream_t stream) {
    const float*         rnn_state = (const float*)d_in[0];
    const float*         ins       = (const float*)d_in[1];
    const unsigned char* resets    = (const unsigned char*)d_in[2];
    const float*         Wi        = (const float*)d_in[3];
    const float*         bi        = (const float*)d_in[4];
    const float*         Whrz      = (const float*)d_in[5];
    const float*         Whn       = (const float*)d_in[6];
    const float*         bhn       = (const float*)d_in[7];

    float* out     = (float*)d_out;
    float* final_h = out;
    float* ys      = out + (size_t)B_DIM * H_DIM;

    __bf16* pwWi = (__bf16*)d_ws;
    __bf16* pwWh = pwWi + W_HALFS;

    pack_weights_kernel<<<512, 256, 0, stream>>>(Wi, Whrz, Whn, pwWi, pwWh);

    const dim3 grid(H_DIM / 16, B_DIM / 16);
    for (int t = 0; t < T_STEPS; ++t) {
        const float* x_t = ins + (size_t)t * B_DIM * D_DIM;
        const unsigned char* rst = resets + (size_t)t * B_DIM;
        const float* h_prev =
            (t == 0) ? rnn_state : (ys + (size_t)(t - 1) * B_DIM * H_DIM);
        float* h_out = ys + (size_t)t * B_DIM * H_DIM;
        gru_step_kernel<<<grid, 32, 0, stream>>>(
            x_t, rst, h_prev, bi, bhn, pwWi, pwWh, h_out,
            (t == T_STEPS - 1) ? final_h : nullptr);
    }
}
